// LSTM_5205500362969
// MI455X (gfx1250) — matrix operand order repaired and checked by reading, not compiled, not run on hardware
//
#include <hip/hip_runtime.h>
#include <hip/hip_bf16.h>

// ---------------------------------------------------------------------------
// Persistent WMMA LSTM for MI455X (gfx1250, wave32).
//   gates = [x_t | h_{t-1}] (64x1024, f16) @ W^T (2048x1024, f16)  -> f32
//   32 workgroups, each owns 16 hidden units (64 gate columns).
//   Weights for the slice live in LDS (128 KB) for the whole kernel.
//   Per step: stage A in LDS, v_wmma_f32_16x16x32_f16, activations,
//   publish h via L2, device-scope atomic grid barrier.
// ---------------------------------------------------------------------------

typedef __attribute__((ext_vector_type(16))) _Float16 v16h;
typedef __attribute__((ext_vector_type(8)))  float    v8f;

constexpr int SEQ   = 2048;
constexpr int BATCH = 64;
constexpr int DIN   = 512;
constexpr int H     = 512;
constexpr int GATES = 4 * H;        // 2048
constexpr int KTOT  = DIN + H;      // 1024
constexpr int NWG   = 32;           // workgroups (grid barrier participants)
constexpr int NTHR  = 256;          // 8 waves of 32

constexpr size_t SBH = (size_t)SEQ * BATCH * H;   // 67,108,864 elements

// workspace layout (bytes)
constexpr size_t W_OFF    = 0;
constexpr size_t W_BYTES  = (size_t)GATES * KTOT * 2;          // 4 MB f16
constexpr size_t B_OFF    = W_OFF + W_BYTES;                   // combined bias f32
constexpr size_t B_BYTES  = (size_t)GATES * 4;                 // 8 KB
constexpr size_t H_OFF    = B_OFF + B_BYTES;                   // h double buffer f16
constexpr size_t H_BYTES  = (size_t)2 * BATCH * H * 2;         // 128 KB
constexpr size_t CNT_OFF  = H_OFF + H_BYTES;                   // barrier counter
constexpr size_t WS_NEED  = CNT_OFF + 64;

// dynamic LDS layout (bytes)
constexpr unsigned LDSW_OFF = 0;                 // 64 x 1024 f16 weight slice
constexpr unsigned LDSA_OFF = 131072;            // 64 x 1024 f16 activations
constexpr unsigned LDSG_OFF = 262144;            // 64 x 64 f32 gate tile
constexpr unsigned LDSC_OFF = 278528;            // 64 x 16 f32 cell state
constexpr unsigned LDS_BYTES = 282624;

__device__ __forceinline__ float sigf(float x) {
  return 1.0f / (1.0f + __expf(-x));
}

// Load a 32-byte WMMA fragment from LDS as two b128 reads.
__device__ __forceinline__ v16h ld_frag(const unsigned* base, int off0, int off1) {
  uint4 p = *(const uint4*)(base + off0);
  uint4 q = *(const uint4*)(base + off1);
  union { unsigned u[8]; v16h h; } r;
  r.u[0] = p.x; r.u[1] = p.y; r.u[2] = p.z; r.u[3] = p.w;
  r.u[4] = q.x; r.u[5] = q.y; r.u[6] = q.z; r.u[7] = q.w;
  return r.h;
}

// ---------------------------------------------------------------------------
// Prep: fuse [W_ih | W_hh] -> f16, combine biases, zero h buffers, reset bar.
// ---------------------------------------------------------------------------
__global__ void lstm_prep(const float* __restrict__ Wih,
                          const float* __restrict__ Whh,
                          const float* __restrict__ bih,
                          const float* __restrict__ bhh,
                          _Float16* __restrict__ Wf16,
                          float* __restrict__ bias,
                          _Float16* __restrict__ hbuf,
                          unsigned* __restrict__ cnt) {
  unsigned i = blockIdx.x * blockDim.x + threadIdx.x;   // covers 2048*1024
  unsigned col = i >> 10, k = i & 1023;
  float v = (k < (unsigned)DIN) ? Wih[col * DIN + k] : Whh[col * H + (k - DIN)];
  Wf16[i] = (_Float16)v;
  if (i < (unsigned)GATES)       bias[i] = bih[i] + bhh[i];
  if (i < (unsigned)(2 * BATCH * H)) hbuf[i] = (_Float16)0.0f;
  if (i == 0)                    *cnt = 0u;
}

// ---------------------------------------------------------------------------
// Persistent recurrent kernel.
// ---------------------------------------------------------------------------
__global__ __launch_bounds__(NTHR, 1) void lstm_run(
    const float* __restrict__ x,        // [SEQ][BATCH][DIN]
    const _Float16* __restrict__ Wf16,  // [GATES][KTOT]
    const float* __restrict__ bias,     // [GATES]
    _Float16* hbuf,                     // 2 x [BATCH][H]
    unsigned* cnt,                      // grid barrier counter
    float* out) {                       // h_out [S,B,H] ++ c_out [S,B,H]
  extern __shared__ __align__(16) char smem[];
  _Float16* lds_w = (_Float16*)(smem + LDSW_OFF);
  _Float16* lds_a = (_Float16*)(smem + LDSA_OFF);
  float*    lds_g = (float*)(smem + LDSG_OFF);
  float*    lds_c = (float*)(smem + LDSC_OFF);
  const unsigned* ldswU = (const unsigned*)lds_w;
  const unsigned* ldsaU = (const unsigned*)lds_a;

  const int tid = threadIdx.x;
  const int j0  = blockIdx.x * 16;          // owned hidden-slice base

  // ---- load weight slice into LDS once: rows r = gate*16 + c, K-contiguous
  {
    const uint4* src = (const uint4*)Wf16;  // 128 uint4 per 1024-half row
    uint4* dst = (uint4*)lds_w;
    for (int idx = tid; idx < 64 * 128; idx += NTHR) {
      int r = idx >> 7, q = idx & 127;
      int gcol = (r >> 4) * H + j0 + (r & 15);
      dst[r * 128 + q] = src[(size_t)gcol * 128 + q];
    }
  }
  // ---- zero cell state
  for (int e = tid; e < BATCH * 16; e += NTHR) lds_c[e] = 0.0f;

  // ---- per-wave geometry
  const int lane    = tid & 31;
  const int wv      = tid >> 5;
  const int mt      = wv & 3;               // M tile (batch rows /16)
  const int ng      = wv >> 1 >> 1;         // 0..1 -> gate pair
  const int laneLow = lane & 15;
  const int hi      = lane >> 4;
  const int rowA    = (mt * 16 + laneLow) * 512;          // uint offset
  const int rowB0   = ((2 * ng) * 16 + laneLow) * 512;    // uint offset
  const int rowB1   = ((2 * ng + 1) * 16 + laneLow) * 512;
  const int aoff    = hi * 8;
  const int boff    = hi * 8;

  // ---- elementwise geometry + bias preload (registers)
  const int hrow = tid >> 2;
  const int cc0  = (tid & 3) * 4;
  float bI[4], bF[4], bG[4], bO[4];
  #pragma unroll
  for (int ii = 0; ii < 4; ++ii) {
    bI[ii] = bias[0 * H + j0 + cc0 + ii];
    bF[ii] = bias[1 * H + j0 + cc0 + ii];
    bG[ii] = bias[2 * H + j0 + cc0 + ii];
    bO[ii] = bias[3 * H + j0 + cc0 + ii];
  }

  for (int t = 0; t < SEQ; ++t) {
    const _Float16* hcur = hbuf + (size_t)(t & 1) * BATCH * H;
    _Float16* hnext      = hbuf + (size_t)((t + 1) & 1) * BATCH * H;

    // ---- stage A = [x_t (f32->f16) | h (f16)] into LDS, K-contiguous rows
    {
      const float4* xt = (const float4*)(x + (size_t)t * BATCH * DIN);
      unsigned* dstU = (unsigned*)lds_a;
      for (int idx = tid; idx < BATCH * DIN / 4; idx += NTHR) {
        float4 v = xt[idx];
        union { _Float16 h[2]; unsigned u; } p0, p1;
        p0.h[0] = (_Float16)v.x; p0.h[1] = (_Float16)v.y;
        p1.h[0] = (_Float16)v.z; p1.h[1] = (_Float16)v.w;
        int r = idx >> 7, c2 = (idx & 127) * 2;         // 512 f32 per row
        dstU[r * 512 + c2]     = p0.u;
        dstU[r * 512 + c2 + 1] = p1.u;
      }
      const uint4* hs = (const uint4*)hcur;             // 64 uint4 per 512-half row
      uint4* dst4 = (uint4*)lds_a;
      for (int idx = tid; idx < BATCH * H / 8; idx += NTHR) {
        int r = idx >> 6, q = idx & 63;
        dst4[r * 128 + 64 + q] = hs[idx];
      }
    }
    __syncthreads();

    // ---- GEMM: 2 accumulator tiles per wave, K = 1024 in 32 steps
    v8f acc0 = {}, acc1 = {};
    #pragma unroll 4
    for (int kt = 0; kt < 32; ++kt) {
      int kd = kt * 16;                                  // uint offset of K chunk
      v16h a  = ld_frag(ldsaU, rowA  + kd + aoff, rowA  + kd + aoff + 4);
      v16h b0 = ld_frag(ldswU, rowB0 + kd + boff, rowB0 + kd + boff + 4);
      v16h b1 = ld_frag(ldswU, rowB1 + kd + boff, rowB1 + kd + boff + 4);
      acc0 = __builtin_amdgcn_wmma_f32_16x16x32_f16(false, a, false, b0,
                                                    (short)0, acc0, false, false);
      acc1 = __builtin_amdgcn_wmma_f32_16x16x32_f16(false, a, false, b1,
                                                    (short)0, acc1, false, false);
    }

    // ---- spill gate tiles to LDS (C/D layout: row = r + hi*8, col = laneLow)
    {
      int grow  = mt * 16 + hi * 8;
      int gcol0 = (2 * ng) * 16 + laneLow;
      #pragma unroll
      for (int r = 0; r < 8; ++r) {
        lds_g[(grow + r) * 64 + gcol0]      = acc0[r];
        lds_g[(grow + r) * 64 + gcol0 + 16] = acc1[r];
      }
    }
    __syncthreads();

    // ---- activations + state update: 4 hidden elems per thread
    #pragma unroll
    for (int ii = 0; ii < 4; ++ii) {
      int cc = cc0 + ii;
      float gi = lds_g[hrow * 64 +  0 + cc] + bI[ii];
      float gf = lds_g[hrow * 64 + 16 + cc] + bF[ii];
      float gg = lds_g[hrow * 64 + 32 + cc] + bG[ii];
      float go = lds_g[hrow * 64 + 48 + cc] + bO[ii];
      float cold = lds_c[hrow * 16 + cc];
      float cn = sigf(gf) * cold + sigf(gi) * tanhf(gg);
      float hn = sigf(go) * tanhf(cn);
      lds_c[hrow * 16 + cc] = cn;
      hnext[hrow * H + j0 + cc] = (_Float16)hn;
      size_t oo = (size_t)t * BATCH * H + (size_t)hrow * H + j0 + cc;
      out[oo]       = hn;   // h_out
      out[SBH + oo] = cn;   // c_out
    }

    // ---- device-wide barrier (monotonic counter, reset each launch by prep)
    __syncthreads();
    if (tid == 0) {
      __threadfence();
      __hip_atomic_fetch_add(cnt, 1u, __ATOMIC_ACQ_REL, __HIP_MEMORY_SCOPE_AGENT);
      unsigned target = (unsigned)(t + 1) * NWG;
      while (__hip_atomic_load(cnt, __ATOMIC_ACQUIRE, __HIP_MEMORY_SCOPE_AGENT) < target) {
        __builtin_amdgcn_s_sleep(1);
      }
    }
    __syncthreads();
  }
}

// ---------------------------------------------------------------------------
extern "C" void kernel_launch(void* const* d_in, const int* in_sizes, int n_in,
                              void* d_out, int out_size, void* d_ws, size_t ws_size,
                              hipStream_t stream) {
  const float* x   = (const float*)d_in[0];
  const float* Wih = (const float*)d_in[1];
  const float* Whh = (const float*)d_in[2];
  const float* bih = (const float*)d_in[3];
  const float* bhh = (const float*)d_in[4];
  float* out = (float*)d_out;

  char* ws = (char*)d_ws;
  _Float16* Wf16 = (_Float16*)(ws + W_OFF);
  float*    bias = (float*)(ws + B_OFF);
  _Float16* hbuf = (_Float16*)(ws + H_OFF);
  unsigned* cnt  = (unsigned*)(ws + CNT_OFF);

  (void)in_sizes; (void)n_in; (void)out_size; (void)ws_size;

  // prep: one thread per weight element (2048*1024 / 256 = 8192 blocks)
  lstm_prep<<<(GATES * KTOT) / NTHR, NTHR, 0, stream>>>(
      Wih, Whh, bih, bhh, Wf16, bias, hbuf, cnt);

  // persistent recurrent kernel: 32 WGs, 276 KB dynamic LDS each
  lstm_run<<<NWG, NTHR, LDS_BYTES, stream>>>(x, Wf16, bias, hbuf, cnt, out);
}
